// Encoder_46746424049961
// MI455X (gfx1250) — matrix operand order repaired and checked by reading, not compiled, not run on hardware
//
#include <hip/hip_runtime.h>
#include <math.h>

// ---------------------------------------------------------------------------
// Transformer encoder layer for MI455X (gfx1250), bf16 WMMA everywhere.
//   x:[8,1024,1024] f32 -> out f32 same shape
// ---------------------------------------------------------------------------

#define D_MODEL 1024
#define HID     4096
#define MROWS   8192      // 8*1024 tokens
#define NHEAD   16
#define DHEAD   64
#define SEQ     1024
#define BATCH   8

typedef __attribute__((ext_vector_type(16))) __bf16 v16bf;
typedef __attribute__((ext_vector_type(8)))  float  v8f;

union Frag { v16bf v; uint4 u[2]; };

__device__ __forceinline__ unsigned short f2bf(float f) {
  unsigned int u = __float_as_uint(f);
  u += 0x7FFFu + ((u >> 16) & 1u);          // round-to-nearest-even
  return (unsigned short)(u >> 16);
}

__device__ __forceinline__ float rmax16(float v) {
#pragma unroll
  for (int m = 1; m < 16; m <<= 1) v = fmaxf(v, __shfl_xor(v, m, 16));
  return v;
}
__device__ __forceinline__ float rsum16(float v) {
#pragma unroll
  for (int m = 1; m < 16; m <<= 1) v += __shfl_xor(v, m, 16);
  return v;
}

// ---------------------------------------------------------------------------
// Pack a f32 weight matrix W[K][N] (row-major, [in,out]) into WMMA B-fragment
// order: fragment(kt,nt) covers K rows kt*32..+31, N cols nt*16..+15.
// Fragment = 512 bf16 (1KB). Lane L owns halves [16L,16L+16):
//   seg = L>>4, n = nt*16 + (L&15), half hh -> W[kt*32 + seg*16 + hh][n].
// ---------------------------------------------------------------------------
__global__ __launch_bounds__(256)
void pack_w(const float* __restrict__ W, unsigned short* __restrict__ out,
            int K, int N, int nfrags) {
  int t    = blockIdx.x * 256 + threadIdx.x;
  int lane = t & 31;
  int frag = t >> 5;
  if (frag >= nfrags) return;
  int nfrag_n = N >> 4;
  int kt  = frag / nfrag_n;
  int nt  = frag % nfrag_n;
  int seg = lane >> 4;
  int n   = (nt << 4) + (lane & 15);
  unsigned short* o = out + (size_t)frag * 512 + lane * 16;
  const float* w = W + (size_t)(kt * 32 + seg * 16) * N + n;
#pragma unroll
  for (int hh = 0; hh < 16; ++hh) o[hh] = f2bf(w[(size_t)hh * N]);
}

// ---------------------------------------------------------------------------
// LayerNorm (ddof=1 to match torch .var()) over last dim (1024), f32 -> bf16
// One block (256 thr) per row.
// ---------------------------------------------------------------------------
__global__ __launch_bounds__(256)
void layernorm_bf16(const float* __restrict__ X, const float* __restrict__ alpha,
                    const float* __restrict__ beta, unsigned short* __restrict__ Y) {
  __shared__ float sbuf[16];
  const int row  = blockIdx.x;
  const int lane = threadIdx.x & 31;
  const int wv   = threadIdx.x >> 5;
  const float* xr = X + (size_t)row * D_MODEL;

  float vals[4];
  float s = 0.f, ss = 0.f;
#pragma unroll
  for (int i = 0; i < 4; ++i) {
    float v = xr[threadIdx.x + i * 256];
    vals[i] = v; s += v; ss += v * v;
  }
#pragma unroll
  for (int m = 1; m < 32; m <<= 1) { s += __shfl_xor(s, m, 32); ss += __shfl_xor(ss, m, 32); }
  if (lane == 0) { sbuf[wv] = s; sbuf[8 + wv] = ss; }
  __syncthreads();
  if (threadIdx.x == 0) {
    float ts = 0.f, tss = 0.f;
    for (int w = 0; w < 8; ++w) { ts += sbuf[w]; tss += sbuf[8 + w]; }
    sbuf[0] = ts; sbuf[8] = tss;
  }
  __syncthreads();
  const float mu   = sbuf[0] * (1.0f / D_MODEL);
  const float var  = (sbuf[8] - sbuf[0] * mu) * (1.0f / (D_MODEL - 1));
  const float rstd = rsqrtf(var + 1e-6f);
  unsigned short* yr = Y + (size_t)row * D_MODEL;
#pragma unroll
  for (int i = 0; i < 4; ++i) {
    int c = threadIdx.x + i * 256;
    yr[c] = f2bf((vals[i] - mu) * rstd * alpha[c] + beta[c]);
  }
}

// ---------------------------------------------------------------------------
// Tiled bf16 GEMM:  D[M=8192, N] = A[M,K] @ Bpacked[K,N] + bias.
// 256 threads = 8 waves; WG tile 128x128; wave computes 16x128 (8 acc tiles).
// B tile (8KB/k-step) staged in LDS via async copies (ASYNCcnt), double
// buffered; per-wave A fragment double-buffered in registers. The k-loop is
// unrolled by 2 so all buffer indices are compile-time constants (keeps the
// A fragments out of scratch/LDS-promoted allocas).
// Epilogue modes:
//   0: Q/K  -> bf16 [b,h,p,64]       1: V -> bf16 [b,h,64,p] (transposed)
//   2: f32 out = res + acc           3: bf16 out = gelu_erf(acc)
//   4: f32 out = res + acc  (final)
// ---------------------------------------------------------------------------
__global__ __launch_bounds__(256)
void gemm_wmma(const unsigned short* __restrict__ A,
               const unsigned short* __restrict__ Bp,
               const float* __restrict__ bias,
               int K, int N, int mode,
               const float* __restrict__ res,
               unsigned short* __restrict__ obf,
               float* __restrict__ of) {
  __shared__ __align__(16) unsigned short bstage[2][4096];   // 2 x 8KB
  const int lane  = threadIdx.x & 31;
  const int wv    = threadIdx.x >> 5;
  const int half  = lane >> 4;
  const int l15   = lane & 15;
  const int mbase = blockIdx.x * 128 + wv * 16;
  const int nbase = blockIdx.y * 128;
  const int nfrag = N >> 4;
  const int ksteps = K >> 5;               // always even here (32 or 128)

  v8f acc[8];
#pragma unroll
  for (int nt = 0; nt < 8; ++nt) {
    float bv = bias[nbase + nt * 16 + l15];
#pragma unroll
    for (int j = 0; j < 8; ++j) acc[nt][j] = bv;
  }

  // async copy of one k-step B tile (8 contiguous packed fragments = 8KB):
  // thread t moves 32 contiguous bytes with two async b128s.
  const unsigned lbase0 = (unsigned)(size_t)&bstage[0][(size_t)threadIdx.x * 16];
  const unsigned lbase1 = (unsigned)(size_t)&bstage[1][(size_t)threadIdx.x * 16];
  auto issue_copy = [&](unsigned loff, int kt) {
    const unsigned short* bsrc =
        Bp + ((size_t)kt * nfrag + (size_t)blockIdx.y * 8) * 512 +
        (size_t)threadIdx.x * 16;
    unsigned long long ga = (unsigned long long)(size_t)bsrc;
    asm volatile("global_load_async_to_lds_b128 %0, %1, off\n\t"
                 "global_load_async_to_lds_b128 %0, %1, off offset:16"
                 :: "v"(loff), "v"(ga) : "memory");
  };
  auto load_a = [&](Frag& a, const unsigned short* arow, int kt) {
    const unsigned short* ap = arow + kt * 32 + half * 16;
    a.u[0] = *(const uint4*)(ap);
    a.u[1] = *(const uint4*)(ap + 8);
  };
  auto compute_step = [&](const Frag& a, int buf) {
    Frag b[8];
#pragma unroll
    for (int nt = 0; nt < 8; ++nt) {
      const unsigned short* bp = &bstage[buf][nt * 512 + lane * 16];
      b[nt].u[0] = *(const uint4*)(bp);
      b[nt].u[1] = *(const uint4*)(bp + 8);
    }
#pragma unroll
    for (int nt = 0; nt < 8; ++nt) {
      acc[nt] = __builtin_amdgcn_wmma_f32_16x16x32_bf16(
          false, a.v, false, b[nt].v, (short)0, acc[nt], false, false);
    }
  };

  const unsigned short* arow = A + (size_t)(mbase + l15) * K;
  Frag a0, a1;
  issue_copy(lbase0, 0);                      // prologue: k-step 0
  load_a(a0, arow, 0);

  for (int kt = 0; kt < ksteps; kt += 2) {
    // ---- even step: consume buf0/a0, prefetch buf1/a1 for kt+1
    issue_copy(lbase1, kt + 1);
    load_a(a1, arow, kt + 1);
    asm volatile("s_wait_asynccnt 0x2" ::: "memory");  // buf0 landed
    __syncthreads();
    compute_step(a0, 0);
    __syncthreads();                          // all done reading buf0

    // ---- odd step: consume buf1/a1, prefetch buf0/a0 for kt+2
    const int knext = (kt + 2 < ksteps) ? kt + 2 : kt + 1;  // clamped (dead)
    issue_copy(lbase0, knext);
    load_a(a0, arow, knext);
    asm volatile("s_wait_asynccnt 0x2" ::: "memory");  // buf1 landed
    __syncthreads();
    compute_step(a1, 1);
    __syncthreads();                          // all done reading buf1
  }

#pragma unroll
  for (int nt = 0; nt < 8; ++nt) {
    const int n = nbase + nt * 16 + l15;
#pragma unroll
    for (int j = 0; j < 8; ++j) {
      const int m = mbase + half * 8 + j;
      float v = acc[nt][j];
      if (mode == 0) {                         // Q/K: [b,h,p,dh] bf16
        int b = m >> 10, p = m & 1023, h = n >> 6, dh = n & 63;
        obf[(((size_t)(b * NHEAD + h)) * SEQ + p) * DHEAD + dh] = f2bf(v);
      } else if (mode == 1) {                  // V: [b,h,dh,p] bf16
        int b = m >> 10, p = m & 1023, h = n >> 6, dh = n & 63;
        obf[(((size_t)(b * NHEAD + h)) * DHEAD + dh) * SEQ + p] = f2bf(v);
      } else if (mode == 2 || mode == 4) {     // residual add, f32 out
        size_t idx = (size_t)m * N + n;
        of[idx] = res[idx] + v;
      } else {                                 // mode 3: exact-erf GELU, bf16
        float g = 0.5f * v * (1.0f + erff(v * 0.70710678118654752f));
        obf[(size_t)m * N + n] = f2bf(g);
      }
    }
  }
}

// ---------------------------------------------------------------------------
// Flash attention, one wave per 16 query rows. Q,K: [b,h,p,64] bf16;
// Vt: [b,h,64,p] bf16; output: [b,p,1024] bf16 (natural layout for O-proj).
// Block = 128 threads (4 waves); grid = (B*H, SEQ/64).
// All K and V fragments for a 32-key block are loaded up front (single
// loadcnt wait), then 4 score WMMAs, online softmax, 4 P@V WMMAs.
// ---------------------------------------------------------------------------
__global__ __launch_bounds__(128)
void attn_flash(const unsigned short* __restrict__ Q,
                const unsigned short* __restrict__ Kb,
                const unsigned short* __restrict__ Vt,
                unsigned short* __restrict__ O) {
  __shared__ __align__(16) unsigned short plds[4][512];   // 16x32 P per wave
  const int lane  = threadIdx.x & 31;
  const int wv    = threadIdx.x >> 5;
  const int half  = lane >> 4;
  const int l15   = lane & 15;
  const int bh    = blockIdx.x;
  const int qbase = blockIdx.y * 64 + wv * 16;

  // Q as two A fragments (dh 0..31, 32..63)
  const unsigned short* qrow = Q + ((size_t)bh * SEQ + qbase + l15) * DHEAD;
  Frag aq0, aq1;
  aq0.u[0] = *(const uint4*)(qrow + half * 16);
  aq0.u[1] = *(const uint4*)(qrow + half * 16 + 8);
  aq1.u[0] = *(const uint4*)(qrow + 32 + half * 16);
  aq1.u[1] = *(const uint4*)(qrow + 32 + half * 16 + 8);

  v8f o[4];
#pragma unroll
  for (int t = 0; t < 4; ++t)
#pragma unroll
    for (int j = 0; j < 8; ++j) o[t][j] = 0.f;
  float mrow[8], lrow[8];
#pragma unroll
  for (int j = 0; j < 8; ++j) { mrow[j] = -3.0e38f; lrow[j] = 0.f; }

  const unsigned short* kb_base = Kb + (size_t)bh * SEQ * DHEAD;
  const unsigned short* vt_base = Vt + (size_t)bh * DHEAD * SEQ;

  for (int kb = 0; kb < SEQ / 32; ++kb) {
    // ---- batch-load all K (4) and V (4) B-fragments for this key block
    const unsigned short* krow0 = kb_base + (size_t)(kb * 32 + l15) * DHEAD;
    const unsigned short* krow1 = krow0 + 16 * DHEAD;
    Frag kf[4], vf[4];
    kf[0].u[0] = *(const uint4*)(krow0 + half * 16);
    kf[0].u[1] = *(const uint4*)(krow0 + half * 16 + 8);
    kf[1].u[0] = *(const uint4*)(krow0 + 32 + half * 16);
    kf[1].u[1] = *(const uint4*)(krow0 + 32 + half * 16 + 8);
    kf[2].u[0] = *(const uint4*)(krow1 + half * 16);
    kf[2].u[1] = *(const uint4*)(krow1 + half * 16 + 8);
    kf[3].u[0] = *(const uint4*)(krow1 + 32 + half * 16);
    kf[3].u[1] = *(const uint4*)(krow1 + 32 + half * 16 + 8);
#pragma unroll
    for (int t = 0; t < 4; ++t) {
      const unsigned short* vrow =
          vt_base + (size_t)(t * 16 + l15) * SEQ + kb * 32;
      vf[t].u[0] = *(const uint4*)(vrow + half * 16);
      vf[t].u[1] = *(const uint4*)(vrow + half * 16 + 8);
    }

    // ---- scores: two 16x16 tiles over 32 keys, K-dim = dh (64 = 2 steps)
    v8f s0, s1;
#pragma unroll
    for (int j = 0; j < 8; ++j) { s0[j] = 0.f; s1[j] = 0.f; }
    s0 = __builtin_amdgcn_wmma_f32_16x16x32_bf16(false, aq0.v, false, kf[0].v, (short)0, s0, false, false);
    s0 = __builtin_amdgcn_wmma_f32_16x16x32_bf16(false, aq1.v, false, kf[1].v, (short)0, s0, false, false);
    s1 = __builtin_amdgcn_wmma_f32_16x16x32_bf16(false, aq0.v, false, kf[2].v, (short)0, s1, false, false);
    s1 = __builtin_amdgcn_wmma_f32_16x16x32_bf16(false, aq1.v, false, kf[3].v, (short)0, s1, false, false);

    // ---- online softmax (rows of D tile live in this lane's 16-lane group)
    float alpha[8];
#pragma unroll
    for (int j = 0; j < 8; ++j) {
      float a = s0[j] * 0.125f;   // 1/sqrt(64)
      float b = s1[j] * 0.125f;
      float mx   = rmax16(fmaxf(a, b));
      float mnew = fmaxf(mrow[j], mx);
      alpha[j]   = __expf(mrow[j] - mnew);
      float p0   = __expf(a - mnew);
      float p1   = __expf(b - mnew);
      float rs   = rsum16(p0 + p1);
      lrow[j] = lrow[j] * alpha[j] + rs;
      mrow[j] = mnew;
      int r = half * 8 + j;                       // P tile row (q-local)
      plds[wv][r * 32 + l15]      = f2bf(p0);     // keys 0..15 of block
      plds[wv][r * 32 + 16 + l15] = f2bf(p1);     // keys 16..31 of block
    }
#pragma unroll
    for (int t = 0; t < 4; ++t)
#pragma unroll
      for (int j = 0; j < 8; ++j) o[t][j] *= alpha[j];

    asm volatile("s_wait_dscnt 0x0" ::: "memory");  // P stores -> A-frag loads

    // ---- P (16x32) as A fragment from LDS
    Frag pa;
    const unsigned short* pr = &plds[wv][l15 * 32];
    pa.u[0] = *(const uint4*)(pr + half * 16);
    pa.u[1] = *(const uint4*)(pr + half * 16 + 8);

    // ---- O += P @ V : 4 dh-tiles (V fragments already in registers)
#pragma unroll
    for (int t = 0; t < 4; ++t) {
      o[t] = __builtin_amdgcn_wmma_f32_16x16x32_bf16(
          false, pa.v, false, vf[t].v, (short)0, o[t], false, false);
    }
  }

  // ---- epilogue: normalize, write [b, p, h*64+dh] bf16
  const int b = bh >> 4, h = bh & 15;
#pragma unroll
  for (int j = 0; j < 8; ++j) {
    const float inv = 1.0f / lrow[j];
    const int r = qbase + half * 8 + j;
#pragma unroll
    for (int t = 0; t < 4; ++t) {
      O[((size_t)b * SEQ + r) * D_MODEL + h * DHEAD + t * 16 + l15] =
          f2bf(o[t][j] * inv);
    }
  }
}

// ---------------------------------------------------------------------------
// Host-side orchestration
// ---------------------------------------------------------------------------
extern "C" void kernel_launch(void* const* d_in, const int* in_sizes, int n_in,
                              void* d_out, int out_size, void* d_ws, size_t ws_size,
                              hipStream_t stream) {
  (void)in_sizes; (void)n_in; (void)out_size; (void)ws_size;
  const float* x   = (const float*)d_in[0];
  const float* a1  = (const float*)d_in[1];
  const float* be1 = (const float*)d_in[2];
  const float* a2  = (const float*)d_in[3];
  const float* be2 = (const float*)d_in[4];
  const float* wq  = (const float*)d_in[5];  const float* bq = (const float*)d_in[6];
  const float* wk  = (const float*)d_in[7];  const float* bk = (const float*)d_in[8];
  const float* wv  = (const float*)d_in[9];  const float* bv = (const float*)d_in[10];
  const float* wo  = (const float*)d_in[11]; const float* bo = (const float*)d_in[12];
  const float* w1  = (const float*)d_in[13]; const float* b1 = (const float*)d_in[14];
  const float* w2  = (const float*)d_in[15]; const float* b2 = (const float*)d_in[16];

  char* ws = (char*)d_ws;
  size_t off = 0;
  auto carve = [&](size_t bytes) -> void* {
    void* p = ws + off; off += (bytes + 255) & ~(size_t)255; return p;
  };
  const size_t MB = 1024 * 1024;
  unsigned short* wqp  = (unsigned short*)carve(2 * MB);
  unsigned short* wkp  = (unsigned short*)carve(2 * MB);
  unsigned short* wvp  = (unsigned short*)carve(2 * MB);
  unsigned short* wop  = (unsigned short*)carve(2 * MB);
  unsigned short* w1p  = (unsigned short*)carve(8 * MB);
  unsigned short* w2p  = (unsigned short*)carve(8 * MB);
  unsigned short* xn   = (unsigned short*)carve(16 * MB);   // ln1(x) bf16
  unsigned short* qb   = (unsigned short*)carve(16 * MB);   // [b,h,p,64]
  unsigned short* kbuf = (unsigned short*)carve(16 * MB);   // [b,h,p,64]
  unsigned short* vt   = (unsigned short*)carve(16 * MB);   // [b,h,64,p]
  unsigned short* atn  = (unsigned short*)carve(16 * MB);   // [b,p,1024]
  float*          x1   = (float*)carve(32 * MB);            // residual 1 (f32)
  unsigned short* h2   = (unsigned short*)carve(16 * MB);   // ln2(x1) bf16
  unsigned short* g    = (unsigned short*)carve(64 * MB);   // gelu out [8192,4096]

  auto pack = [&](const float* W, unsigned short* out, int K, int N) {
    int frags = (K >> 5) * (N >> 4);
    int threads = frags * 32;
    pack_w<<<(threads + 255) / 256, 256, 0, stream>>>(W, out, K, N, frags);
  };
  pack(wq, wqp, D_MODEL, D_MODEL);
  pack(wk, wkp, D_MODEL, D_MODEL);
  pack(wv, wvp, D_MODEL, D_MODEL);
  pack(wo, wop, D_MODEL, D_MODEL);
  pack(w1, w1p, D_MODEL, HID);
  pack(w2, w2p, HID, D_MODEL);

  layernorm_bf16<<<MROWS, 256, 0, stream>>>(x, a1, be1, xn);

  dim3 gd(MROWS / 128, D_MODEL / 128);
  gemm_wmma<<<gd, 256, 0, stream>>>(xn, wqp, bq, D_MODEL, D_MODEL, 0, nullptr, qb,   nullptr);
  gemm_wmma<<<gd, 256, 0, stream>>>(xn, wkp, bk, D_MODEL, D_MODEL, 0, nullptr, kbuf, nullptr);
  gemm_wmma<<<gd, 256, 0, stream>>>(xn, wvp, bv, D_MODEL, D_MODEL, 1, nullptr, vt,   nullptr);

  attn_flash<<<dim3(BATCH * NHEAD, SEQ / 64), 128, 0, stream>>>(qb, kbuf, vt, atn);

  gemm_wmma<<<gd, 256, 0, stream>>>(atn, wop, bo, D_MODEL, D_MODEL, 2, x, nullptr, x1);

  layernorm_bf16<<<MROWS, 256, 0, stream>>>(x1, a2, be2, h2);

  dim3 gd1(MROWS / 128, HID / 128);
  gemm_wmma<<<gd1, 256, 0, stream>>>(h2, w1p, b1, D_MODEL, HID, 3, nullptr, g, nullptr);

  gemm_wmma<<<gd, 256, 0, stream>>>(g, w2p, b2, HID, D_MODEL, 4, x1, nullptr, (float*)d_out);
}
